// GPT2Attention_55568286876207
// MI455X (gfx1250) — matrix operand order repaired and checked by reading, not compiled, not run on hardware
//
#include <hip/hip_runtime.h>
#include <stdint.h>

#define EMBED   1024
#define HEADS   16
#define HDIM    64
#define SEQ     2048
#define BATCH   4
#define NTOK    (BATCH*SEQ)   /* 8192 */
#define KTILES  (EMBED/32)    /* 32 */

typedef __attribute__((ext_vector_type(16))) __bf16 v16bf;
typedef __attribute__((ext_vector_type(8)))  float  v8f;
typedef __attribute__((ext_vector_type(4)))  int    vi4;

union FragB { v16bf v; uint4 q[2]; unsigned short s[16]; };
union AccF  { v8f v; float f[8]; };

#if __has_builtin(__builtin_amdgcn_global_load_async_to_lds_b128)
#define HAVE_ASYNC 1
#else
#define HAVE_ASYNC 0
#endif

#if HAVE_ASYNC
#if __has_builtin(__builtin_amdgcn_s_wait_asynccnt)
#define WAIT_ASYNC(n) __builtin_amdgcn_s_wait_asynccnt(n)
#else
#define WAIT_ASYNC(n) asm volatile("s_wait_asynccnt %0" :: "n"(n) : "memory")
#endif
#else
#define WAIT_ASYNC(n) ((void)0)
#endif

// 16-byte tile-chunk copy global->LDS; async (ASYNCcnt) on CDNA5, sync fallback.
__device__ __forceinline__ void cp16(void* l, const void* g) {
#if HAVE_ASYNC
    __builtin_amdgcn_global_load_async_to_lds_b128(
        (__attribute__((address_space(1))) vi4*)(void*)g,
        (__attribute__((address_space(3))) vi4*)l, 0, 0);
#else
    *(uint4*)l = *(const uint4*)g;
#endif
}

__device__ __forceinline__ unsigned short f2bf(float x) {
    union { float f; unsigned u; } a; a.f = x;
    unsigned r = a.u + 0x7FFFu + ((a.u >> 16) & 1u);
    return (unsigned short)(r >> 16);
}

__global__ void cvt4_f32_bf16(const float* __restrict__ src,
                              unsigned short* __restrict__ dst, int n4) {
    int i = blockIdx.x * blockDim.x + threadIdx.x;
    if (i < n4) {
        float4 f = ((const float4*)src)[i];
        ushort4 h;
        h.x = f2bf(f.x); h.y = f2bf(f.y); h.z = f2bf(f.z); h.w = f2bf(f.w);
        ((ushort4*)dst)[i] = h;
    }
}

// out[n][k] = bf16(in[k][n]); in: [K][N] f32, out: [N][K] bf16 (LDS-tiled transpose)
__global__ __launch_bounds__(256)
void transpose_cvt(const float* __restrict__ in, unsigned short* __restrict__ outp,
                   int K, int N) {
    __shared__ float tile[32][33];
    const int kb = blockIdx.y * 32, nb = blockIdx.x * 32;
    const int tx = threadIdx.x & 31, ty = threadIdx.x >> 5;
#pragma unroll
    for (int j = ty; j < 32; j += 8)
        tile[j][tx] = in[(size_t)(kb + j) * N + nb + tx];
    __syncthreads();
#pragma unroll
    for (int j = ty; j < 32; j += 8)
        outp[(size_t)(nb + j) * K + kb + tx] = f2bf(tile[tx][j]);
}

// per-head transpose: Vb [B,H,S,D] bf16 -> Vtg [B,H,D,S] bf16 (64x64 LDS tiles)
__global__ __launch_bounds__(256)
void transpose_v(const unsigned short* __restrict__ Vb, unsigned short* __restrict__ Vtg) {
    __shared__ __align__(16) unsigned short tl[64][72];
    const int s0 = blockIdx.x * 64;
    const size_t head = (size_t)blockIdx.y * SEQ * HDIM;
    const int t = threadIdx.x;
#pragma unroll
    for (int p = 0; p < 2; ++p) {
        int c = p * 256 + t;
        int s = c >> 3, dk = (c & 7) * 8;
        *(uint4*)(&tl[s][dk]) = *(const uint4*)(Vb + head + (size_t)(s0 + s) * HDIM + dk);
    }
    __syncthreads();
#pragma unroll
    for (int p = 0; p < 2; ++p) {
        int c = p * 256 + t;
        int d = c >> 3, k8 = (c & 7) * 8;
        unsigned short tmp[8];
#pragma unroll
        for (int j = 0; j < 8; ++j) tmp[j] = tl[k8 + j][d];
        *(uint4*)(Vtg + head + (size_t)d * SEQ + s0 + k8) = *(uint4*)tmp;
    }
}

// A: bf16 [8192,1024] row-major. Bt: bf16 [N,1024] (pre-transposed weights).
// MODE 0: N=3072, scatter bf16 q/k/v into [B,H,S,D] (+bias)
// MODE 1: N=1024, f32 out [8192,1024] (+bias)
template<int MODE>
__global__ __launch_bounds__(256)
void gemm_wmma(const unsigned short* __restrict__ Abf, const unsigned short* __restrict__ Bt,
               const float* __restrict__ bias, unsigned short* __restrict__ Qo,
               unsigned short* __restrict__ Ko, unsigned short* __restrict__ Vo,
               float* __restrict__ out)
{
    __shared__ __align__(16) unsigned short As[2][128][32];  // [m][k]
    __shared__ __align__(16) unsigned short Bs[2][128][32];  // [n][k]
    const int t = threadIdx.x;
    const int lane = t & 31, wave = t >> 5;
    const int waveM = wave >> 1, waveN = wave & 1;
    const int m0 = blockIdx.y * 128;
    const int n0 = blockIdx.x * 128;
    const int l16 = lane & 15, hi = lane >> 4;

    AccF acc[2][4];
#pragma unroll
    for (int mi = 0; mi < 2; ++mi)
#pragma unroll
        for (int ni = 0; ni < 4; ++ni)
#pragma unroll
            for (int v = 0; v < 8; ++v) acc[mi][ni].f[v] = 0.0f;

    // per-thread: 2 chunks of A + 2 chunks of B (16B each) per k-tile
    auto stage = [&](int buf, int k0) {
#pragma unroll
        for (int j = 0; j < 2; ++j) {
            int idx = t * 2 + j;
            int r = idx >> 2, ko = (idx & 3) * 8;
            cp16(&As[buf][r][ko], Abf + (size_t)(m0 + r) * EMBED + k0 + ko);
            cp16(&Bs[buf][r][ko], Bt + (size_t)(n0 + r) * EMBED + k0 + ko);
        }
    };

    stage(0, 0);
    for (int it = 0; it < KTILES; ++it) {
        const int buf = it & 1;
        __syncthreads();                       // prev compute done -> buf^1 reusable
        if (it + 1 < KTILES) { stage(buf ^ 1, (it + 1) * 32); WAIT_ASYNC(4); }
        else                 { WAIT_ASYNC(0); }
        __syncthreads();                       // tile `it` visible to all waves

        FragB a[2], b[4];
#pragma unroll
        for (int mi = 0; mi < 2; ++mi) {
            int row = waveM * 32 + mi * 16 + l16;
            int kb = hi * 8;
            a[mi].q[0] = *(const uint4*)(&As[buf][row][kb]);
            a[mi].q[1] = *(const uint4*)(&As[buf][row][kb + 16]);
        }
#pragma unroll
        for (int ni = 0; ni < 4; ++ni) {
            int nrow = waveN * 64 + ni * 16 + l16;
            int kb = hi * 8;
            b[ni].q[0] = *(const uint4*)(&Bs[buf][nrow][kb]);
            b[ni].q[1] = *(const uint4*)(&Bs[buf][nrow][kb + 16]);
        }
#pragma unroll
        for (int ni = 0; ni < 4; ++ni)
#pragma unroll
            for (int mi = 0; mi < 2; ++mi)
                acc[mi][ni].v = __builtin_amdgcn_wmma_f32_16x16x32_bf16(
                    false, a[mi].v, false, b[ni].v, (short)0, acc[mi][ni].v, false, false);
    }
    // ---- epilogue ----
#pragma unroll
    for (int mi = 0; mi < 2; ++mi)
#pragma unroll
        for (int ni = 0; ni < 4; ++ni)
#pragma unroll
            for (int v = 0; v < 8; ++v) {
                int g   = m0 + waveM * 32 + mi * 16 + hi * 8 + v;   // token row
                int col = n0 + waveN * 64 + ni * 16 + l16;
                float val = acc[mi][ni].f[v] + bias[col];
                if (MODE == 0) {
                    int which = col >> 10, rem = col & 1023;
                    int hh = rem >> 6, d = rem & 63;
                    int bb = g >> 11, ss = g & 2047;
                    size_t idx = ((size_t)(bb * HEADS + hh) * SEQ + ss) * HDIM + d;
                    unsigned short hv = f2bf(val);
                    if (which == 0) Qo[idx] = hv;
                    else if (which == 1) Ko[idx] = hv;
                    else Vo[idx] = hv;
                } else {
                    out[(size_t)g * EMBED + col] = val;
                }
            }
}

// Flash attention: grid (S/64, B*H), 128 threads (4 waves x 16 query rows).
// K in [B,H,S,D]; V pre-transposed in [B,H,D,S] -> all staging is straight async copies,
// double-buffered so block kb+1 streams in while block kb computes.
__global__ __launch_bounds__(128)
void attn_wmma(const unsigned short* __restrict__ Q, const unsigned short* __restrict__ K,
               const unsigned short* __restrict__ Vtg, unsigned short* __restrict__ O)
{
    __shared__ __align__(16) unsigned short Kbuf[2][64][64];   // [key][d]
    __shared__ __align__(16) unsigned short Vt[2][64][64];     // [d][key]
    __shared__ __align__(16) unsigned short Pbuf[4][16][64];
    const int t = threadIdx.x, lane = t & 31, wave = t >> 5;
    const int l16 = lane & 15, hi = lane >> 4;
    const int qb = blockIdx.x, bh = blockIdx.y;
    const int bb = bh >> 4, hh = bh & 15;
    const size_t head = (size_t)bh * SEQ * HDIM;
    const int qRow0 = qb * 64 + wave * 16;
    const unsigned short* kp0 = K + head;
    const unsigned short* vp0 = Vtg + head;

    auto stage = [&](int buf, int kb) {
#pragma unroll
        for (int p = 0; p < 4; ++p) {
            int e = (p * 128 + t) * 8;
            int r = e >> 6, c = e & 63;
            cp16(&Kbuf[buf][r][c], kp0 + (size_t)kb * 64 * HDIM + e);
            cp16(&Vt[buf][r][c],   vp0 + (size_t)r * SEQ + kb * 64 + c);
        }
    };

    FragB aQ[2];
    {
        const unsigned short* qp = Q + head + (size_t)(qRow0 + l16) * HDIM;
#pragma unroll
        for (int ks = 0; ks < 2; ++ks) {
            int d0 = ks * 32 + hi * 8;
            aQ[ks].q[0] = *(const uint4*)(qp + d0);
            aQ[ks].q[1] = *(const uint4*)(qp + d0 + 16);
        }
    }
    AccF o[4];
    float mrow[8], lrow[8];
#pragma unroll
    for (int v = 0; v < 8; ++v) { mrow[v] = -3.0e38f; lrow[v] = 0.0f; }
#pragma unroll
    for (int ns = 0; ns < 4; ++ns)
#pragma unroll
        for (int v = 0; v < 8; ++v) o[ns].f[v] = 0.0f;

    stage(0, 0);
    for (int kb = 0; kb <= qb; ++kb) {
        const int buf = kb & 1;
        __syncthreads();                        // prev compute done -> buf^1 reusable
        if (kb < qb) { stage(buf ^ 1, kb + 1); WAIT_ASYNC(8); }
        else         { WAIT_ASYNC(0); }
        __syncthreads();                        // block kb visible to all waves

        // ---- scores: S16x64 = Q(16x64) * K^T ----
        FragB bk[4][2];
#pragma unroll
        for (int ns = 0; ns < 4; ++ns) {
            int key = ns * 16 + l16;
#pragma unroll
            for (int ks = 0; ks < 2; ++ks) {
                int d0 = ks * 32 + hi * 8;
                bk[ns][ks].q[0] = *(const uint4*)(&Kbuf[buf][key][d0]);
                bk[ns][ks].q[1] = *(const uint4*)(&Kbuf[buf][key][d0 + 16]);
            }
        }
        AccF sc[4];
#pragma unroll
        for (int ns = 0; ns < 4; ++ns) {
            v8f z = {};
            z = __builtin_amdgcn_wmma_f32_16x16x32_bf16(
                false, aQ[0].v, false, bk[ns][0].v, (short)0, z, false, false);
            z = __builtin_amdgcn_wmma_f32_16x16x32_bf16(
                false, aQ[1].v, false, bk[ns][1].v, (short)0, z, false, false);
            sc[ns].v = z;
        }
        // ---- causal mask + scale + online softmax ----
        float tmax[8];
#pragma unroll
        for (int v = 0; v < 8; ++v) tmax[v] = -3.0e38f;
#pragma unroll
        for (int ns = 0; ns < 4; ++ns)
#pragma unroll
            for (int v = 0; v < 8; ++v) {
                int keyg = kb * 64 + ns * 16 + l16;
                int sg   = qRow0 + hi * 8 + v;
                float val = (keyg <= sg) ? sc[ns].f[v] * 0.125f : -10000.0f;
                sc[ns].f[v] = val;
                tmax[v] = fmaxf(tmax[v], val);
            }
#pragma unroll
        for (int msk = 1; msk < 16; msk <<= 1)
#pragma unroll
            for (int v = 0; v < 8; ++v)
                tmax[v] = fmaxf(tmax[v], __shfl_xor(tmax[v], msk));
        float rs[8];
#pragma unroll
        for (int v = 0; v < 8; ++v) {
            float nm = fmaxf(mrow[v], tmax[v]);
            float sca = __expf(mrow[v] - nm);
            mrow[v] = nm;
            lrow[v] *= sca;
            rs[v] = 0.0f;
#pragma unroll
            for (int nt = 0; nt < 4; ++nt) o[nt].f[v] *= sca;
        }
#pragma unroll
        for (int ns = 0; ns < 4; ++ns)
#pragma unroll
            for (int v = 0; v < 8; ++v) {
                float p = __expf(sc[ns].f[v] - mrow[v]);
                rs[v] += p;
                Pbuf[wave][hi * 8 + v][ns * 16 + l16] = f2bf(p);
            }
#pragma unroll
        for (int msk = 1; msk < 16; msk <<= 1)
#pragma unroll
            for (int v = 0; v < 8; ++v) rs[v] += __shfl_xor(rs[v], msk);
#pragma unroll
        for (int v = 0; v < 8; ++v) lrow[v] += rs[v];
        // cross-lane LDS dependency within the wave: drain DS ops explicitly
        asm volatile("s_wait_dscnt 0" ::: "memory");
        // ---- O += P(16x64) * V(64x64) ----
        FragB aP[2], bv[4][2];
#pragma unroll
        for (int ks = 0; ks < 2; ++ks) {
            int c0 = ks * 32 + hi * 8;
            aP[ks].q[0] = *(const uint4*)(&Pbuf[wave][l16][c0]);
            aP[ks].q[1] = *(const uint4*)(&Pbuf[wave][l16][c0 + 16]);
        }
#pragma unroll
        for (int ns = 0; ns < 4; ++ns) {
            int d = ns * 16 + l16;
#pragma unroll
            for (int ks = 0; ks < 2; ++ks) {
                int k0 = ks * 32 + hi * 8;
                bv[ns][ks].q[0] = *(const uint4*)(&Vt[buf][d][k0]);
                bv[ns][ks].q[1] = *(const uint4*)(&Vt[buf][d][k0 + 16]);
            }
        }
#pragma unroll
        for (int ns = 0; ns < 4; ++ns)
#pragma unroll
            for (int ks = 0; ks < 2; ++ks)
                o[ns].v = __builtin_amdgcn_wmma_f32_16x16x32_bf16(
                    false, aP[ks].v, false, bv[ns][ks].v, (short)0, o[ns].v, false, false);
    }
    // ---- normalize + write merged-head bf16 O[B,S,E] ----
#pragma unroll
    for (int v = 0; v < 8; ++v) {
        float inv = 1.0f / lrow[v];
        int ss = qRow0 + hi * 8 + v;
        size_t rowbase = ((size_t)(bb * SEQ + ss)) * EMBED + hh * HDIM;
#pragma unroll
        for (int ns = 0; ns < 4; ++ns)
            O[rowbase + ns * 16 + l16] = f2bf(o[ns].f[v] * inv);
    }
}

extern "C" void kernel_launch(void* const* d_in, const int* in_sizes, int n_in,
                              void* d_out, int out_size, void* d_ws, size_t ws_size,
                              hipStream_t stream) {
    (void)in_sizes; (void)n_in; (void)out_size; (void)ws_size;
    const float* X     = (const float*)d_in[0];
    const float* Wqkv  = (const float*)d_in[1];
    const float* bqkv  = (const float*)d_in[2];
    const float* Wproj = (const float*)d_in[3];
    const float* bproj = (const float*)d_in[4];
    float* out = (float*)d_out;

    size_t off = 0;
    char* ws = (char*)d_ws;
    auto take = [&](size_t nbytes) -> void* {
        void* p = ws + off;
        off += (nbytes + 255) & ~(size_t)255;
        return p;
    };
    unsigned short* Xbf     = (unsigned short*)take((size_t)NTOK * EMBED * 2);
    unsigned short* WqkvT   = (unsigned short*)take((size_t)EMBED * 3 * EMBED * 2);
    unsigned short* WprojT  = (unsigned short*)take((size_t)EMBED * EMBED * 2);
    unsigned short* Qb  = (unsigned short*)take((size_t)NTOK * EMBED * 2);
    unsigned short* Kb  = (unsigned short*)take((size_t)NTOK * EMBED * 2);
    unsigned short* Vb  = (unsigned short*)take((size_t)NTOK * EMBED * 2);
    unsigned short* Vtg = (unsigned short*)take((size_t)NTOK * EMBED * 2);
    unsigned short* Ob  = (unsigned short*)take((size_t)NTOK * EMBED * 2);

    // pre-passes: X -> bf16 (straight), W -> bf16 transposed [N][K]
    {
        int n4 = (NTOK * EMBED) / 4;
        cvt4_f32_bf16<<<n4 / 256, 256, 0, stream>>>(X, Xbf, n4);
        transpose_cvt<<<dim3(3 * EMBED / 32, EMBED / 32), 256, 0, stream>>>(
            Wqkv, WqkvT, EMBED, 3 * EMBED);
        transpose_cvt<<<dim3(EMBED / 32, EMBED / 32), 256, 0, stream>>>(
            Wproj, WprojT, EMBED, EMBED);
    }
    // QKV projection (writes q/k/v bf16 in [B,H,S,D])
    gemm_wmma<0><<<dim3(3 * EMBED / 128, NTOK / 128), 256, 0, stream>>>(
        Xbf, WqkvT, bqkv, Qb, Kb, Vb, nullptr);
    // V -> [B,H,D,S] so attention V staging is a straight async copy
    transpose_v<<<dim3(SEQ / 64, BATCH * HEADS), 256, 0, stream>>>(Vb, Vtg);
    // flash attention (double-buffered async staging)
    attn_wmma<<<dim3(SEQ / 64, BATCH * HEADS), 128, 0, stream>>>(Qb, Kb, Vtg, Ob);
    // output projection
    gemm_wmma<1><<<dim3(EMBED / 128, NTOK / 128), 256, 0, stream>>>(
        Ob, WprojT, bproj, nullptr, nullptr, nullptr, out);
}
